// minGRUCell_7902739824854
// MI455X (gfx1250) — matrix operand order repaired and checked by reading, not compiled, not run on hardware
//
#include <hip/hip_runtime.h>

#define B_DIM 4
#define S_DIM 8192
#define D_DIM 512
#define H_DIM 512
#define M_DIM (B_DIM * S_DIM)      // 32768 rows of x viewed as (M, D)
#define CHUNK 128
#define NCHUNK (S_DIM / CHUNK)     // 64
#define NSEQ (B_DIM * H_DIM)       // 2048 independent scan lanes
#define EPS_F 1e-8f

#define BLOCK_M 128                // 8 waves * 16 rows
#define BLOCK_N 64                 // 4 n-tiles of 16
#define KSTEP 32
#define NKSTEP (D_DIM / KSTEP)     // 16

// LDS buffer layout (in halfs), per ping-pong buffer:
//   [0,2048)    Wz tile   64 x 32
//   [2048,4096) Wh tile   64 x 32
//   [4096,8192) x  tile  128 x 32
#define LDS_WZ 0
#define LDS_WH 2048
#define LDS_X  4096
#define LDS_BUF 8192               // halfs per buffer (16 KB)

typedef __attribute__((ext_vector_type(16))) _Float16 v16h;
typedef __attribute__((ext_vector_type(8)))  _Float16 v8h;
typedef __attribute__((ext_vector_type(8)))  float    v8f;
typedef __attribute__((ext_vector_type(4)))  unsigned int u32x4;
typedef __attribute__((ext_vector_type(8)))  int      i32x8;
typedef __attribute__((ext_vector_type(4)))  int      i32x4;

#if __has_builtin(__builtin_amdgcn_tensor_load_to_lds)
#define HAVE_TDM 1
#else
#define HAVE_TDM 0
#endif

// ---------------------------------------------------------------------------
// f32 -> f16 pre-pass (8 elements / thread, vectorized)
// ---------------------------------------------------------------------------
__global__ __launch_bounds__(256)
void convert_f16(const float* __restrict__ src, _Float16* __restrict__ dst, int n8)
{
    const int i = blockIdx.x * 256 + threadIdx.x;
    if (i >= n8) return;
    const float4* s = (const float4*)src + (size_t)i * 2;
    float4 a = s[0], b = s[1];
    v8h o;
    o[0] = (_Float16)a.x; o[1] = (_Float16)a.y; o[2] = (_Float16)a.z; o[3] = (_Float16)a.w;
    o[4] = (_Float16)b.x; o[5] = (_Float16)b.y; o[6] = (_Float16)b.z; o[7] = (_Float16)b.w;
    ((v8h*)dst)[i] = o;
}

#if HAVE_TDM
// ---------------------------------------------------------------------------
// Issue one TDM 2-D tile load (f16 elements): tile = 32 (k, contiguous) x
// tile_rows, row stride = 512 elements. D# packing per CDNA5 ISA §8.3/§8.4.
// ---------------------------------------------------------------------------
__device__ __forceinline__ void tdm_load_tile_f16(const _Float16* gptr,
                                                  unsigned lds_byte,
                                                  unsigned tensor_rows,
                                                  unsigned tile_rows)
{
    unsigned long long ga = (unsigned long long)(const void*)gptr;
    u32x4 g0;
    g0[0] = 1u;                                          // count=1 (valid user D#)
    g0[1] = lds_byte;                                    // lds_addr (bytes)
    g0[2] = (unsigned)(ga & 0xffffffffu);                // global_addr[31:0]
    g0[3] = (unsigned)((ga >> 32) & 0x01ffffffu)         // global_addr[56:32]
          | (2u << 30);                                  // type = 2 ("image")
    i32x8 g1;
    g1[0] = (int)(1u << 16);                             // wg_mask=0, data_size=1 (2B)
    g1[1] = (int)((D_DIM & 0xffffu) << 16);              // tensor_dim0[15:0]=512
    g1[2] = (int)((tensor_rows & 0xffffu) << 16);        // dim0 hi=0 | tensor_dim1 lo16
    g1[3] = (int)(((tensor_rows >> 16) & 0xffffu)        // tensor_dim1 hi16
          | ((unsigned)KSTEP << 16));                    // tile_dim0 = 32
    g1[4] = (int)tile_rows;                              // tile_dim1; tile_dim2=0
    g1[5] = D_DIM;                                       // tensor_dim0_stride = 512
    g1[6] = 0;                                           // stride hi / dim1_stride lo
    g1[7] = 0;
    i32x4 z4 = {0, 0, 0, 0};
#if defined(__clang_major__) && (__clang_major__ >= 23)
    i32x8 z8 = {0, 0, 0, 0, 0, 0, 0, 0};
    __builtin_amdgcn_tensor_load_to_lds(g0, g1, z4, z4, z8, 0);
#else
    __builtin_amdgcn_tensor_load_to_lds(g0, g1, z4, z4, 0);
#endif
}
#endif

// ---------------------------------------------------------------------------
// Fused dual-GEMM (f16 WMMA, f32 acc) + activation epilogue.
// a_ws   = clip(1 - sigmoid(x Wz^T + bz), eps, 1-eps)
// bv_out = sigmoid(.) * tanh(x Wh^T + bh)
// ---------------------------------------------------------------------------
__global__ __launch_bounds__(256)
void mingru_gemm_act(const _Float16* __restrict__ xh,
                     const _Float16* __restrict__ wzh,
                     const _Float16* __restrict__ whh,
                     const float* __restrict__ bz,
                     const float* __restrict__ bh,
                     float* __restrict__ a_ws,
                     float* __restrict__ bv_out)
{
    const int tid  = threadIdx.x;
    const int lane = tid & 31;
    const int wave = tid >> 5;
    const int n0 = blockIdx.x * BLOCK_N;
    const int m0 = blockIdx.y * BLOCK_M + wave * 16;

    v8f zero = {0.f, 0.f, 0.f, 0.f, 0.f, 0.f, 0.f, 0.f};
    v8f accz[4], acch[4];
#pragma unroll
    for (int t = 0; t < 4; ++t) { accz[t] = zero; acch[t] = zero; }

    // A fragment addressing (contiguous k = 16h + s, same order as B):
    // lanes 0-15 hold K {0..15}; lanes 16-31 hold K {16..31}
    const int akh = (lane >> 4) * 16;
    // B fragment: lane n holds 16 contiguous K of its column
    const int bkq = (lane >> 4) * 16;
    const int bnl = lane & 15;

#if HAVE_TDM
    __shared__ __align__(128) _Float16 smem[2 * LDS_BUF];   // 32 KB, ping-pong

    // Per-producer-wave TDM parameters (waves 0,1,2 issue; all 8 compute)
    const _Float16* tsrc =
        (wave == 0) ? wzh + (size_t)n0 * D_DIM :
        (wave == 1) ? whh + (size_t)n0 * D_DIM :
                      xh + (size_t)(blockIdx.y * BLOCK_M) * D_DIM;
    const unsigned tregion = (wave == 0) ? LDS_WZ : (wave == 1) ? LDS_WH : LDS_X;
    const unsigned ttrows  = (wave == 2) ? (unsigned)M_DIM : (unsigned)H_DIM;
    const unsigned ttile   = (wave == 2) ? 128u : 64u;

    // Prologue: k-chunk 0 into buffer 0
    if (wave < 3)
        tdm_load_tile_f16(tsrc, (unsigned)(size_t)(void*)&smem[tregion],
                          ttrows, ttile);

    for (int ki = 0; ki < NKSTEP; ++ki) {
        const int buf = ki & 1;
        if (wave < 3) {
            if (ki + 1 < NKSTEP) {
                tdm_load_tile_f16(tsrc + (ki + 1) * KSTEP,
                                  (unsigned)(size_t)(void*)&smem[(buf ^ 1) * LDS_BUF + tregion],
                                  ttrows, ttile);
                __builtin_amdgcn_s_wait_tensorcnt(1);   // chunk ki complete
            } else {
                __builtin_amdgcn_s_wait_tensorcnt(0);
            }
        }
        __syncthreads();                                 // tile visible to all waves

        const _Float16* Lb = &smem[buf * LDS_BUF];
        v16h afrag;
        {
            const _Float16* Ax = &Lb[LDS_X + (wave * 16 + (lane & 15)) * KSTEP + akh];
            v8h alo = *(const v8h*)Ax;
            v8h ahi = *(const v8h*)(Ax + 8);
#pragma unroll
            for (int i = 0; i < 8; ++i) { afrag[i] = alo[i]; afrag[8 + i] = ahi[i]; }
        }
#pragma unroll
        for (int t = 0; t < 4; ++t) {
            v16h bfz = *(const v16h*)&Lb[LDS_WZ + (t * 16 + bnl) * KSTEP + bkq];
            v16h bfh = *(const v16h*)&Lb[LDS_WH + (t * 16 + bnl) * KSTEP + bkq];
            accz[t] = __builtin_amdgcn_wmma_f32_16x16x32_f16(
                false, afrag, false, bfz, (short)0, accz[t], false, false);
            acch[t] = __builtin_amdgcn_wmma_f32_16x16x32_f16(
                false, afrag, false, bfh, (short)0, acch[t], false, false);
        }
        __syncthreads();                                 // reads done before reuse
    }
#else
    // Fallback: direct global f16 fragment loads (W stays hot in 192MB L2)
    const _Float16* xrow = xh + (size_t)(m0 + (lane & 15)) * D_DIM + akh;
    for (int ki = 0; ki < NKSTEP; ++ki) {
        const int k0 = ki * KSTEP;
        v16h afrag;
        {
            v8h alo = *(const v8h*)(xrow + k0);
            v8h ahi = *(const v8h*)(xrow + k0 + 8);
#pragma unroll
            for (int i = 0; i < 8; ++i) { afrag[i] = alo[i]; afrag[8 + i] = ahi[i]; }
        }
#pragma unroll
        for (int t = 0; t < 4; ++t) {
            const size_t nrow = (size_t)(n0 + t * 16 + bnl) * D_DIM + k0 + bkq;
            v16h bfz = *(const v16h*)(wzh + nrow);
            v16h bfh = *(const v16h*)(whh + nrow);
            accz[t] = __builtin_amdgcn_wmma_f32_16x16x32_f16(
                false, afrag, false, bfz, (short)0, accz[t], false, false);
            acch[t] = __builtin_amdgcn_wmma_f32_16x16x32_f16(
                false, afrag, false, bfh, (short)0, acch[t], false, false);
        }
    }
#endif

    // Epilogue: C/D layout — VGPR r, lane L -> row m0 + r + 8*(L>>4), col (L&15)
    const int mrow_base = m0 + 8 * (lane >> 4);
#pragma unroll
    for (int t = 0; t < 4; ++t) {
        const int n = n0 + t * 16 + (lane & 15);
        const float bzv = bz[n];
        const float bhv = bh[n];
#pragma unroll
        for (int r = 0; r < 8; ++r) {
            const int m = mrow_base + r;
            const float z  = 1.0f / (1.0f + __expf(-(accz[t][r] + bzv)));
            const float th = tanhf(acch[t][r] + bhv);
            const float a  = fminf(fmaxf(1.0f - z, EPS_F), 1.0f - EPS_F);
            const size_t idx = (size_t)m * H_DIM + n;
            a_ws[idx]   = a;
            bv_out[idx] = z * th;
        }
    }
}

// ---------------------------------------------------------------------------
// Phase 1 of chunked affine scan: per-(seq,chunk) composed affine (A, B).
// agg layout: [chunk][seq] -> all accesses coalesced.
// ---------------------------------------------------------------------------
__global__ __launch_bounds__(256)
void mingru_chunk_reduce(const float* __restrict__ a_ws,
                         const float* __restrict__ bv,
                         float* __restrict__ aggA,
                         float* __restrict__ aggB)
{
    const int g   = blockIdx.x * 256 + threadIdx.x;  // [0, NSEQ*NCHUNK)
    const int c   = g >> 11;                         // / NSEQ
    const int seq = g & (NSEQ - 1);
    const int b   = seq >> 9;                        // / H_DIM
    const int n   = seq & (H_DIM - 1);
    size_t idx = ((size_t)b * S_DIM + (size_t)c * CHUNK) * H_DIM + n;
    float A = 1.0f, Bc = 0.0f;
    for (int s = 0; s < CHUNK; ++s) {
        const float a = a_ws[idx];
        const float v = bv[idx];
        Bc = fmaf(a, Bc, v);
        A *= a;
        idx += H_DIM;
    }
    aggA[g] = A;
    aggB[g] = Bc;
}

// ---------------------------------------------------------------------------
// Phase 2: serial scan over 64 chunk aggregates per sequence (2048 lanes).
// In-place converts aggB to the exclusive carry-in per chunk.
// ---------------------------------------------------------------------------
__global__ __launch_bounds__(256)
void mingru_seq_scan(const float* __restrict__ aggA, float* __restrict__ aggB)
{
    const int seq = blockIdx.x * 256 + threadIdx.x;  // [0, NSEQ)
    float carry = 0.0f;                              // h_0 = 0
    for (int c = 0; c < NCHUNK; ++c) {
        const int i = c * NSEQ + seq;
        const float A  = aggA[i];
        const float Bc = aggB[i];
        aggB[i] = carry;                             // carry-in for chunk c
        carry = fmaf(A, carry, Bc);
    }
}

// ---------------------------------------------------------------------------
// Phase 3: apply carry-in within each chunk, overwriting bv with h in place.
// ---------------------------------------------------------------------------
__global__ __launch_bounds__(256)
void mingru_apply(const float* __restrict__ a_ws,
                  const float* __restrict__ carryIn,
                  float* __restrict__ out)
{
    const int g   = blockIdx.x * 256 + threadIdx.x;
    const int c   = g >> 11;
    const int seq = g & (NSEQ - 1);
    const int b   = seq >> 9;
    const int n   = seq & (H_DIM - 1);
    size_t idx = ((size_t)b * S_DIM + (size_t)c * CHUNK) * H_DIM + n;
    float h = carryIn[g];
    for (int s = 0; s < CHUNK; ++s) {
        h = fmaf(a_ws[idx], h, out[idx]);
        out[idx] = h;
        idx += H_DIM;
    }
}

// ---------------------------------------------------------------------------
extern "C" void kernel_launch(void* const* d_in, const int* in_sizes, int n_in,
                              void* d_out, int out_size, void* d_ws, size_t ws_size,
                              hipStream_t stream)
{
    const float* x  = (const float*)d_in[0];
    const float* Wz = (const float*)d_in[1];
    const float* bz = (const float*)d_in[2];
    const float* Wh = (const float*)d_in[3];
    const float* bh = (const float*)d_in[4];
    float* out = (float*)d_out;

    // Workspace layout:
    //   a_ws : M*H f32          (64 MB)
    //   aggA : NSEQ*NCHUNK f32  (512 KB)
    //   aggB : NSEQ*NCHUNK f32  (512 KB)
    //   xh   : M*D f16          (32 MB)
    //   wzh  : H*D f16          (512 KB)
    //   whh  : H*D f16          (512 KB)
    char* ws = (char*)d_ws;
    float* a_ws = (float*)ws;
    float* aggA = (float*)(ws + (size_t)M_DIM * H_DIM * 4);
    float* aggB = aggA + (size_t)NSEQ * NCHUNK;
    _Float16* xh  = (_Float16*)(ws + (size_t)M_DIM * H_DIM * 4
                                   + (size_t)NSEQ * NCHUNK * 8);
    _Float16* wzh = xh + (size_t)M_DIM * D_DIM;
    _Float16* whh = wzh + (size_t)H_DIM * D_DIM;

    // Pre-pass: one-time f32 -> f16 of GEMM operands
    const int nx8 = (M_DIM * D_DIM) / 8;
    const int nw8 = (H_DIM * D_DIM) / 8;
    convert_f16<<<(nx8 + 255) / 256, 256, 0, stream>>>(x,  xh,  nx8);
    convert_f16<<<(nw8 + 255) / 256, 256, 0, stream>>>(Wz, wzh, nw8);
    convert_f16<<<(nw8 + 255) / 256, 256, 0, stream>>>(Wh, whh, nw8);

    dim3 ggrid(H_DIM / BLOCK_N, M_DIM / BLOCK_M, 1);   // (8, 256)
    mingru_gemm_act<<<ggrid, 256, 0, stream>>>(xh, wzh, whh, bz, bh, a_ws, out);

    mingru_chunk_reduce<<<(NSEQ * NCHUNK) / 256, 256, 0, stream>>>(a_ws, out, aggA, aggB);
    mingru_seq_scan<<<NSEQ / 256, 256, 0, stream>>>(aggA, aggB);
    mingru_apply<<<(NSEQ * NCHUNK) / 256, 256, 0, stream>>>(a_ws, aggB, out);
}
